// Transformer_model_28905129902181
// MI455X (gfx1250) — matrix operand order repaired and checked by reading, not compiled, not run on hardware
//
#include <hip/hip_runtime.h>

typedef __attribute__((ext_vector_type(16))) _Float16 v16h;
typedef __attribute__((ext_vector_type(8)))  _Float16 v8h;
typedef __attribute__((ext_vector_type(8)))  float    v8f;
typedef __attribute__((ext_vector_type(4)))  int      i4v;

#define LAG      168
#define LPAD     176
#define HORIZON  24
#define DMODEL   6
#define FFD      18
#define NLAYER   3
#define NTILES   11      // 176 / 16
#define KCHUNKS  6       // 192 / 32 k-steps for P@V
#define KPAD     32      // padded feature dim for Q/K
#define PSTRIDE  192
#define NTHREADS 128
#define NWAVES   4
#define EPS      1e-5f
#define SCALE    0.4082482904638631f  // 1/sqrt(6)

struct __align__(32) SMem {
  float    h[LAG * DMODEL];            // 4032 B running hidden state (fp32)
  float    o[LAG * DMODEL];            // 4032 B attention output (fp32)
  // ---- zeroed-once region (contiguous, 53248 B) ----
  _Float16 Qh[LPAD * KPAD];            // 11264 B f16 Q, K-padded to 32
  _Float16 Kh[LPAD * KPAD];            // 11264 B
  _Float16 Vt[16 * PSTRIDE];           //  6144 B V transposed [feature][token]
  _Float16 P[NWAVES][16 * PSTRIDE];    // 24576 B per-wave softmax tiles
};                                      // total 61312 B

// A-fragment 16x32 f16 from K-padded row-major [.. ][32]:
// lane m=lane&15 holds row; VGPR0-7 = K(half*16 + 0..15)
__device__ inline v16h load_a32(const _Float16* row_p, int half) {
  const v8h lo = *(const v8h*)(row_p + half * 16);
  const v8h hi = *(const v8h*)(row_p + half * 16 + 8);
  v16h a;
#pragma unroll
  for (int e = 0; e < 8; ++e) { a[e] = lo[e]; a[8 + e] = hi[e]; }
  return a;
}

// B-fragment 32x16 f16: per lane, elements 0..15 = K(half*16 + 0..15),
// contiguous in our layouts -> one 32-byte LDS load (2x ds_load_b128)
__device__ inline v16h load_b_cont(const _Float16* p) {
  return *(const v16h*)p;
}

__global__ __launch_bounds__(NTHREADS)
void tiny_transformer_kernel(
    const float* __restrict__ input,     const float* __restrict__ pos_emb,
    const float* __restrict__ in_proj_w, const float* __restrict__ in_proj_b,
    const float* __restrict__ out_w,     const float* __restrict__ out_b,
    const float* __restrict__ ln1_g,     const float* __restrict__ ln1_b,
    const float* __restrict__ lin1_w,    const float* __restrict__ lin1_b,
    const float* __restrict__ lin2_w,    const float* __restrict__ lin2_b,
    const float* __restrict__ ln2_g,     const float* __restrict__ ln2_b,
    const float* __restrict__ head_w,    const float* __restrict__ head_b,
    float* __restrict__ out)
{
  __shared__ SMem sm;
  const int b    = blockIdx.x;
  const int tid  = threadIdx.x;
  const int lane = tid & 31;
  const int wid  = tid >> 5;

  // ---- zero the padded f16 region once (Qh,Kh,Vt,P are contiguous) -------
  {
    i4v* z = (i4v*)sm.Qh;
    const int n16 = (int)((sizeof(sm.Qh) + sizeof(sm.Kh) + sizeof(sm.Vt) +
                           sizeof(sm.P)) / sizeof(i4v));  // 3328
    const i4v zero = {0, 0, 0, 0};
    for (int i = tid; i < n16; i += NTHREADS) z[i] = zero;
  }

  // ---- Phase 0: embedding x = [input, pos_emb] ----------------------------
  for (int t = tid; t < LAG; t += NTHREADS) {
    sm.h[t * DMODEL + 0] = input[(size_t)b * LAG + t];
#pragma unroll
    for (int d = 0; d < 5; ++d)
      sm.h[t * DMODEL + 1 + d] = pos_emb[t * 5 + d];
  }
  __syncthreads();

  for (int layer = 0; layer < NLAYER; ++layer) {
    // ---- Phase 1: QKV projection (fp32 VALU, store f16) -------------------
    const float* Wqkv = in_proj_w + layer * 3 * DMODEL * DMODEL;
    const float* Bqkv = in_proj_b + layer * 3 * DMODEL;
    for (int t = tid; t < LAG; t += NTHREADS) {
      float hv[DMODEL];
#pragma unroll
      for (int d = 0; d < DMODEL; ++d) hv[d] = sm.h[t * DMODEL + d];
#pragma unroll
      for (int e = 0; e < 3 * DMODEL; ++e) {
        float acc = Bqkv[e];
#pragma unroll
        for (int d = 0; d < DMODEL; ++d) acc += hv[d] * Wqkv[e * DMODEL + d];
        const _Float16 hval = (_Float16)acc;
        if (e < DMODEL)          sm.Qh[t * KPAD + e] = hval;
        else if (e < 2 * DMODEL) sm.Kh[t * KPAD + (e - DMODEL)] = hval;
        else                     sm.Vt[(e - 2 * DMODEL) * PSTRIDE + t] = hval;
      }
    }
    __syncthreads();

    // ---- Phase 2: attention, one row-tile of 16 queries per wave ----------
    _Float16* Pl = sm.P[wid];
    const int m    = lane & 15;
    const int half = lane >> 4;
    for (int ti = wid; ti < NTILES; ti += NWAVES) {
      const int row0 = ti << 4;
      v8f sacc[NTILES];
      const v16h aq = load_a32(sm.Qh + (row0 + m) * KPAD, half);
#pragma unroll
      for (int j = 0; j < NTILES; ++j) {
        // B = K^T: per lane token (16j+m), features half*16..half*16+15
        const v16h bk = load_b_cont(sm.Kh + ((j << 4) + m) * KPAD + (half << 4));
        v8f c = {};
        sacc[j] = __builtin_amdgcn_wmma_f32_16x16x32_f16(
            false, aq, false, bk, (short)0, c, false, false);
      }
      // scale + mask columns >= 168 (only last tile, cols 8..15)
      const bool maskc = m >= (LAG - ((NTILES - 1) << 4));
#pragma unroll
      for (int j = 0; j < NTILES; ++j)
#pragma unroll
        for (int r = 0; r < 8; ++r) {
          float x = sacc[j][r] * SCALE;
          if (j == NTILES - 1 && maskc) x = -1e30f;
          sacc[j][r] = x;
        }
      // row max / sum: each row's 16 cols live in one half-wave per VGPR
      float rmax[8], rinv[8];
#pragma unroll
      for (int r = 0; r < 8; ++r) {
        float mx = -1e30f;
#pragma unroll
        for (int j = 0; j < NTILES; ++j) mx = fmaxf(mx, sacc[j][r]);
#pragma unroll
        for (int s = 1; s < 16; s <<= 1) mx = fmaxf(mx, __shfl_xor(mx, s, 16));
        rmax[r] = mx;
      }
#pragma unroll
      for (int r = 0; r < 8; ++r) {
        float s = 0.0f;
#pragma unroll
        for (int j = 0; j < NTILES; ++j) {
          const float e = __expf(sacc[j][r] - rmax[r]);
          sacc[j][r] = e;
          s += e;
        }
#pragma unroll
        for (int sh = 1; sh < 16; sh <<= 1) s += __shfl_xor(s, sh, 16);
        rinv[r] = 1.0f / s;
      }
      // write P (f16) from C-layout: lane holds (row r+8*half, col m)
#pragma unroll
      for (int r = 0; r < 8; ++r) {
        const int rl = r + (half << 3);
#pragma unroll
        for (int j = 0; j < NTILES; ++j)
          Pl[rl * PSTRIDE + (j << 4) + m] = (_Float16)(sacc[j][r] * rinv[r]);
      }
      // O = P @ V  (K = 192 tokens in 6 chunks of 32)
      v8f oacc = {};
#pragma unroll
      for (int kt = 0; kt < KCHUNKS; ++kt) {
        const v16h ap = load_a32(Pl + m * PSTRIDE + (kt << 5), half);
        // B from Vt: feature m (row), tokens kt*32 + half*16 + 0..15 contiguous
        const v16h bv = load_b_cont(sm.Vt + m * PSTRIDE + (kt << 5) + (half << 4));
        oacc = __builtin_amdgcn_wmma_f32_16x16x32_f16(
            false, ap, false, bv, (short)0, oacc, false, false);
      }
#pragma unroll
      for (int r = 0; r < 8; ++r) {
        const int row = row0 + r + (half << 3);
        if (row < LAG && m < DMODEL) sm.o[row * DMODEL + m] = oacc[r];
      }
    }
    __syncthreads();

    // ---- Phase 3: out-proj + LN1 + FFN + LN2 (fp32 per-token) -------------
    const float* Wo  = out_w  + layer * DMODEL * DMODEL;
    const float* Bo  = out_b  + layer * DMODEL;
    const float* G1  = ln1_g  + layer * DMODEL;
    const float* Be1 = ln1_b  + layer * DMODEL;
    const float* W1  = lin1_w + layer * FFD * DMODEL;
    const float* Bf1 = lin1_b + layer * FFD;
    const float* W2  = lin2_w + layer * DMODEL * FFD;
    const float* Bf2 = lin2_b + layer * DMODEL;
    const float* G2  = ln2_g  + layer * DMODEL;
    const float* Be2 = ln2_b  + layer * DMODEL;
    for (int t = tid; t < LAG; t += NTHREADS) {
      float ov[DMODEL], hv[DMODEL], s1[DMODEL];
#pragma unroll
      for (int d = 0; d < DMODEL; ++d) {
        ov[d] = sm.o[t * DMODEL + d];
        hv[d] = sm.h[t * DMODEL + d];
      }
#pragma unroll
      for (int e = 0; e < DMODEL; ++e) {
        float acc = Bo[e];
#pragma unroll
        for (int d = 0; d < DMODEL; ++d) acc += ov[d] * Wo[e * DMODEL + d];
        s1[e] = hv[e] + acc;
      }
      float mean = 0.f, var = 0.f;
#pragma unroll
      for (int d = 0; d < DMODEL; ++d) mean += s1[d];
      mean *= (1.0f / DMODEL);
#pragma unroll
      for (int d = 0; d < DMODEL; ++d) { const float z = s1[d] - mean; var += z * z; }
      var *= (1.0f / DMODEL);
      float inv = rsqrtf(var + EPS);
      float h1[DMODEL];
#pragma unroll
      for (int d = 0; d < DMODEL; ++d)
        h1[d] = (s1[d] - mean) * inv * G1[d] + Be1[d];
      float f[FFD];
#pragma unroll
      for (int e = 0; e < FFD; ++e) {
        float acc = Bf1[e];
#pragma unroll
        for (int d = 0; d < DMODEL; ++d) acc += h1[d] * W1[e * DMODEL + d];
        f[e] = fmaxf(acc, 0.0f);
      }
      float s2[DMODEL];
#pragma unroll
      for (int d = 0; d < DMODEL; ++d) {
        float acc = Bf2[d];
#pragma unroll
        for (int e = 0; e < FFD; ++e) acc += f[e] * W2[d * FFD + e];
        s2[d] = h1[d] + acc;
      }
      mean = 0.f; var = 0.f;
#pragma unroll
      for (int d = 0; d < DMODEL; ++d) mean += s2[d];
      mean *= (1.0f / DMODEL);
#pragma unroll
      for (int d = 0; d < DMODEL; ++d) { const float z = s2[d] - mean; var += z * z; }
      var *= (1.0f / DMODEL);
      inv = rsqrtf(var + EPS);
#pragma unroll
      for (int d = 0; d < DMODEL; ++d)
        sm.h[t * DMODEL + d] = (s2[d] - mean) * inv * G2[d] + Be2[d];
    }
    __syncthreads();
  }

  // ---- final residual from embedding + relu -------------------------------
  for (int t = tid; t < LAG; t += NTHREADS) {
    const float x0 = input[(size_t)b * LAG + t];
    sm.h[t * DMODEL + 0] = fmaxf(sm.h[t * DMODEL + 0] + x0, 0.0f);
#pragma unroll
    for (int d = 0; d < 5; ++d)
      sm.h[t * DMODEL + 1 + d] =
          fmaxf(sm.h[t * DMODEL + 1 + d] + pos_emb[t * 5 + d], 0.0f);
  }
  __syncthreads();

  // ---- head: out[b][o] = h_flat . head_w[o] + head_b[o] -------------------
  // 96 threads: 24 outputs x 4 partials, reduced with width-4 shuffles.
  if (tid < HORIZON * 4) {
    const int o = tid >> 2, part = tid & 3;
    const int k0 = part * (LAG * DMODEL / 4);         // 1008/4 = 252
    const float* w = head_w + (size_t)o * (LAG * DMODEL);
    float acc = (part == 0) ? head_b[o] : 0.0f;
    for (int k = k0; k < k0 + LAG * DMODEL / 4; ++k) acc += sm.h[k] * w[k];
    acc += __shfl_xor(acc, 1, 4);
    acc += __shfl_xor(acc, 2, 4);
    if (part == 0) out[(size_t)b * HORIZON + o] = acc;
  }
}

extern "C" void kernel_launch(void* const* d_in, const int* in_sizes, int n_in,
                              void* d_out, int out_size, void* d_ws, size_t ws_size,
                              hipStream_t stream) {
  const float* input     = (const float*)d_in[0];
  const float* pos_emb   = (const float*)d_in[1];
  const float* in_proj_w = (const float*)d_in[2];
  const float* in_proj_b = (const float*)d_in[3];
  const float* out_w     = (const float*)d_in[4];
  const float* out_b     = (const float*)d_in[5];
  const float* ln1_g     = (const float*)d_in[6];
  const float* ln1_b     = (const float*)d_in[7];
  const float* lin1_w    = (const float*)d_in[8];
  const float* lin1_b    = (const float*)d_in[9];
  const float* lin2_w    = (const float*)d_in[10];
  const float* lin2_b    = (const float*)d_in[11];
  const float* ln2_g     = (const float*)d_in[12];
  const float* ln2_b     = (const float*)d_in[13];
  const float* head_w    = (const float*)d_in[14];
  const float* head_b    = (const float*)d_in[15];
  float* out = (float*)d_out;

  const int Bsz = in_sizes[0] / LAG;   // 8192
  dim3 grid(Bsz), block(NTHREADS);
  tiny_transformer_kernel<<<grid, block, 0, stream>>>(
      input, pos_emb, in_proj_w, in_proj_b, out_w, out_b,
      ln1_g, ln1_b, lin1_w, lin1_b, lin2_w, lin2_b,
      ln2_g, ln2_b, head_w, head_b, out);
}
